// CrossAttention_81106162417715
// MI455X (gfx1250) — matrix operand order repaired and checked by reading, not compiled, not run on hardware
//
#include <hip/hip_runtime.h>

typedef unsigned short u16;
typedef __attribute__((ext_vector_type(16))) __bf16 v16bf;
typedef __attribute__((ext_vector_type(8)))  __bf16 v8bf;
typedef __attribute__((ext_vector_type(8)))  float  v8f;
typedef __attribute__((ext_vector_type(4)))  unsigned int u32x4;
typedef __attribute__((ext_vector_type(8)))  int i32x8;
typedef __attribute__((ext_vector_type(4)))  int i32x4;

#define N_HEAD   16
#define HEAD_DIM 64
#define EMB      1024
#define DCTX     768
#define BATCH    4
#define SEQ      2048
#define CTX      768

#if __has_builtin(__builtin_amdgcn_tensor_load_to_lds) && __has_builtin(__builtin_amdgcn_s_wait_tensorcnt)
#define HAVE_TDM 1
#else
#define HAVE_TDM 0
#endif

__device__ __forceinline__ u16 f2bf(float f) {
  unsigned int u = __float_as_uint(f);
  u += 0x7FFFu + ((u >> 16) & 1u);   // round-to-nearest-even
  return (u16)(u >> 16);
}

#if HAVE_TDM
// Issue one TDM 2D tile load (bf16 elements): tile_h rows x tile_w elems,
// global row stride = row_stride elems, packed row-major into LDS at lds_off.
// D# packing per CDNA5 ISA sec 8.3/8.4. Must be executed by ONE wave only.
__device__ __forceinline__ void tdm_load_2d(unsigned lds_off, const void* gptr,
                                            unsigned tile_w, unsigned tile_h,
                                            unsigned row_stride) {
  unsigned long long ga = (unsigned long long)gptr;
  u32x4 g0;
  g0[0] = 1u;                                        // count=1 (valid), user mode
  g0[1] = lds_off;                                   // lds_addr (bytes)
  g0[2] = (unsigned)(ga & 0xFFFFFFFFu);              // global_addr[31:0]
  g0[3] = (unsigned)((ga >> 32) & 0x01FFFFFFu)       // global_addr[56:32]
          | 0x80000000u;                             // type=2 ("image")
  i32x8 g1;
  g1[0] = (int)(1u << 16);                           // data_size=1 (2B), wg_mask=0
  g1[1] = (int)((tile_w & 0xFFFFu) << 16);           // tensor_dim0[15:0]
  g1[2] = (int)(((tile_w >> 16) & 0xFFFFu) | ((tile_h & 0xFFFFu) << 16)); // dim0 hi | dim1 lo
  g1[3] = (int)(((tile_h >> 16) & 0xFFFFu) | ((tile_w & 0xFFFFu) << 16)); // dim1 hi | tile_dim0
  g1[4] = (int)(tile_h & 0xFFFFu);                   // tile_dim1 (tile_dim2=0 -> 2D)
  g1[5] = (int)row_stride;                           // tensor_dim0_stride[31:0]
  g1[6] = 0;                                         // stride0 hi | stride1 lo
  g1[7] = 0;                                         // stride1 hi
  i32x4 z4 = {0, 0, 0, 0};
#if defined(__clang_major__) && __clang_major__ >= 23
  i32x8 z8 = {0, 0, 0, 0, 0, 0, 0, 0};
  __builtin_amdgcn_tensor_load_to_lds(g0, g1, z4, z4, z8, 0);
#else
  __builtin_amdgcn_tensor_load_to_lds(g0, g1, z4, z4, 0);
#endif
}
#endif

__global__ void cast_bf16_kernel(const float* __restrict__ in, u16* __restrict__ out, int n) {
  int i = blockIdx.x * blockDim.x + threadIdx.x;
  if (i < n) out[i] = f2bf(in[i]);
}

// Out[M,N] = A[M,K] @ B[K,N] + bias. bf16 in, fp32 accumulate.
// Block tile 256x128, 8 waves (4x2), each wave 64x64 = 4x4 WMMA frags.
// K-step 32, TDM double-buffered: DMA of step t+1 overlaps WMMA of step t.
template <bool OUT_BF16>
__global__ __launch_bounds__(256) void gemm_bias_kernel(
    const u16* __restrict__ A, const u16* __restrict__ Bw,
    const float* __restrict__ bias, void* __restrict__ Out,
    int M, int N, int K) {
  __shared__ __align__(32) u16 As[2][256][32];   // 2 x 16 KB
  __shared__ __align__(32) u16 Bs[2][32][128];   // 2 x 8 KB
  const int tid  = threadIdx.x;
  const int wid  = tid >> 5, lane = tid & 31;
  const int hf   = lane >> 4, l16 = lane & 15;
  const int rowBase = blockIdx.y * 256;
  const int colBase = blockIdx.x * 128;
  const int wRow = (wid >> 1) * 64;   // 0,64,128,192
  const int wCol = (wid & 1) * 64;    // 0,64
  const int T = K >> 5;               // K-steps

  v8f acc[4][4] = {};
#if HAVE_TDM
  if (wid == 0) {   // prologue: DMA step 0 into buffer 0
    tdm_load_2d((unsigned)(unsigned long long)(void*)&As[0][0][0],
                A + (size_t)rowBase * K, 32, 256, (unsigned)K);
    tdm_load_2d((unsigned)(unsigned long long)(void*)&Bs[0][0][0],
                Bw + colBase, 128, 32, (unsigned)N);
  }
#endif
  for (int t = 0; t < T; ++t) {
    const int cur = t & 1;
    const int k0 = t << 5;
#if HAVE_TDM
    if (wid == 0) {
      if (t + 1 < T) {   // DMA step t+1 into the other buffer (its readers done at prev end-barrier)
        tdm_load_2d((unsigned)(unsigned long long)(void*)&As[cur ^ 1][0][0],
                    A + (size_t)rowBase * K + (k0 + 32), 32, 256, (unsigned)K);
        tdm_load_2d((unsigned)(unsigned long long)(void*)&Bs[cur ^ 1][0][0],
                    Bw + (size_t)(k0 + 32) * N + colBase, 128, 32, (unsigned)N);
        __builtin_amdgcn_s_wait_tensorcnt(2);   // in-order: step t's pair complete
      } else {
        __builtin_amdgcn_s_wait_tensorcnt(0);
      }
    }
#else
#pragma unroll
    for (int v = 0; v < 4; ++v) {               // A tile: 1024 x 16B
      int lin = tid + v * 256;
      int r = lin >> 2, c = (lin & 3) * 8;
      *(uint4*)&As[cur][r][c] = *(const uint4*)&A[(size_t)(rowBase + r) * K + k0 + c];
    }
#pragma unroll
    for (int v = 0; v < 2; ++v) {               // B tile: 512 x 16B
      int lin = tid + v * 256;
      int r = lin >> 4, c = (lin & 15) * 8;
      *(uint4*)&Bs[cur][r][c] = *(const uint4*)&Bw[(size_t)(k0 + r) * N + colBase + c];
    }
#endif
    __syncthreads();   // publish buffer `cur`

    v16bf a[4], b[4];
#pragma unroll
    for (int i = 0; i < 4; ++i) {
      // A-frag: lane holds row M, k = base(0/16) + 8*half + i%8
      const u16* ap = &As[cur][wRow + i * 16 + l16][hf * 8];
      v8bf lo = *(const v8bf*)ap;
      v8bf hi = *(const v8bf*)(ap + 16);
#pragma unroll
      for (int e = 0; e < 8; ++e) { a[i][e] = lo[e]; a[i][e + 8] = hi[e]; }
      // B-frag: lane holds col N, k = base(0/16) + 8*half + i%8
      for (int e = 0; e < 8; ++e) { b[i][e] = __builtin_bit_cast(__bf16, Bs[cur][hf * 8 + e][wCol + i * 16 + l16]); b[i][e + 8] = __builtin_bit_cast(__bf16, Bs[cur][16 + hf * 8 + e][wCol + i * 16 + l16]); }
    }
#pragma unroll
    for (int i = 0; i < 4; ++i)
#pragma unroll
      for (int j = 0; j < 4; ++j)
        acc[i][j] = __builtin_amdgcn_wmma_f32_16x16x32_bf16(
            false, a[i], false, b[j], (short)0, acc[i][j], false, false);
    __syncthreads();   // reads of buffer `cur` done; it may be DMA'd for step t+2
  }
  // Epilogue: D layout -> lane holds col (lane&15), VGPR r -> row r + 8*(lane>>4)
#pragma unroll
  for (int j = 0; j < 4; ++j) {
    int col = colBase + wCol + j * 16 + l16;
    float bv = bias[col];
#pragma unroll
    for (int i = 0; i < 4; ++i)
#pragma unroll
      for (int r = 0; r < 8; ++r) {
        size_t row = (size_t)rowBase + wRow + i * 16 + hf * 8 + r;
        float v = acc[i][j][r] + bv;
        if (OUT_BF16) ((u16*)Out)[row * N + col] = f2bf(v);
        else          ((float*)Out)[row * N + col] = v;
      }
  }
}

// Flash attention: one block = 128 query rows for one (b,h). 8 waves, 16 q-rows each.
__global__ __launch_bounds__(256) void attn_kernel(
    const u16* __restrict__ Q, const u16* __restrict__ Kc,
    const u16* __restrict__ Vc, u16* __restrict__ O) {
  __shared__ __align__(32) u16 Qs[128][64];       // [q][d]     16 KB
  __shared__ __align__(32) u16 Kt[64][64];        // [d][kv]     8 KB (transposed)
  __shared__ __align__(32) u16 Vs[64][64];        // [kv][d]     8 KB
  __shared__ __align__(32) u16 Ps[8][16][64];     // per-wave P 16 KB
  const int tid = threadIdx.x, wid = tid >> 5, lane = tid & 31;
  const int hf = lane >> 4, l16 = lane & 15;
  const int b = blockIdx.z, h = blockIdx.y;
  const int q0 = blockIdx.x * 128;
  const float scale = 0.125f;   // 1/sqrt(64)

  const u16* Qg = Q + ((size_t)b * SEQ + q0) * EMB + h * HEAD_DIM;
#if HAVE_TDM
  if (wid == 0)
    tdm_load_2d((unsigned)(unsigned long long)(void*)Qs, Qg, 64, 128, EMB);
#else
#pragma unroll
  for (int v = 0; v < 4; ++v) {
    int lin = tid + v * 256;
    int r = lin >> 3, c = (lin & 7) * 8;
    *(uint4*)&Qs[r][c] = *(const uint4*)&Qg[(size_t)r * EMB + c];
  }
#endif

  float m_i[8], l_i[8];
  v8f o[4] = {};
#pragma unroll
  for (int r = 0; r < 8; ++r) { m_i[r] = -1e30f; l_i[r] = 0.f; }

  for (int c0 = 0; c0 < CTX; c0 += 64) {
    __syncthreads();   // prior-iteration LDS reads done
    const u16* Kg = Kc + ((size_t)b * CTX + c0) * EMB + h * HEAD_DIM;
    const u16* Vg = Vc + ((size_t)b * CTX + c0) * EMB + h * HEAD_DIM;
#if HAVE_TDM
    if (wid == 0)
      tdm_load_2d((unsigned)(unsigned long long)(void*)Vs, Vg, 64, 64, EMB);
#else
#pragma unroll
    for (int v = 0; v < 2; ++v) {
      int lin = tid + v * 256;
      int r = lin >> 3, c = (lin & 7) * 8;
      *(uint4*)&Vs[r][c] = *(const uint4*)&Vg[(size_t)r * EMB + c];
    }
#endif
#pragma unroll
    for (int v = 0; v < 2; ++v) {               // K -> LDS transposed [d][kv]
      int lin = tid + v * 256;
      int r = lin >> 3, c = (lin & 7) * 8;
      union { uint4 q; u16 u[8]; } kd;
      kd.q = *(const uint4*)&Kg[(size_t)r * EMB + c];
#pragma unroll
      for (int e = 0; e < 8; ++e) Kt[c + e][r] = kd.u[e];
    }
#if HAVE_TDM
    if (wid == 0) __builtin_amdgcn_s_wait_tensorcnt(0);  // covers Qs (iter 0) + Vs
#endif
    __syncthreads();

    // S = Q @ K^T : per wave [16 q x 64 kv], k-dim = head_dim (2 steps of 32)
    v8f s[4] = {};
#pragma unroll
    for (int ks = 0; ks < 2; ++ks) {
      const u16* ap = &Qs[wid * 16 + l16][ks * 32 + hf * 8];
      v8bf lo = *(const v8bf*)ap;
      v8bf hi = *(const v8bf*)(ap + 16);
      v16bf a;
#pragma unroll
      for (int e = 0; e < 8; ++e) { a[e] = lo[e]; a[e + 8] = hi[e]; }
#pragma unroll
      for (int n = 0; n < 4; ++n) {
        v16bf bm;
        for (int e = 0; e < 8; ++e) { bm[e] = __builtin_bit_cast(__bf16, Kt[ks * 32 + hf * 8 + e][n * 16 + l16]); bm[e + 8] = __builtin_bit_cast(__bf16, Kt[ks * 32 + 16 + hf * 8 + e][n * 16 + l16]); }
        s[n] = __builtin_amdgcn_wmma_f32_16x16x32_bf16(
            false, a, false, bm, (short)0, s[n], false, false);
      }
    }

    // Online softmax. Row M = r + 8*hf; its 16 cols/frag live across the 16-lane half.
    float mnew[8], alpha[8];
#pragma unroll
    for (int r = 0; r < 8; ++r) {
      float mx = fmaxf(fmaxf(s[0][r], s[1][r]), fmaxf(s[2][r], s[3][r]));
#pragma unroll
      for (int off = 8; off >= 1; off >>= 1)
        mx = fmaxf(mx, __shfl_xor(mx, off, 32));
      mx *= scale;
      mnew[r]  = fmaxf(m_i[r], mx);
      alpha[r] = __expf(m_i[r] - mnew[r]);
      m_i[r]   = mnew[r];
    }
#pragma unroll
    for (int r = 0; r < 8; ++r) {
      float sum = 0.f;
#pragma unroll
      for (int n = 0; n < 4; ++n) {
        float p = __expf(s[n][r] * scale - mnew[r]);
        sum += p;
        Ps[wid][r + 8 * hf][n * 16 + l16] = f2bf(p);
      }
#pragma unroll
      for (int off = 8; off >= 1; off >>= 1)
        sum += __shfl_xor(sum, off, 32);
      l_i[r] = l_i[r] * alpha[r] + sum;
#pragma unroll
      for (int n = 0; n < 4; ++n) o[n][r] *= alpha[r];
    }

    // O += P @ V (k-dim = 64 kv, 2 steps of 32); P re-read in A-frag layout via LDS
#pragma unroll
    for (int ks = 0; ks < 2; ++ks) {
      const u16* ap = &Ps[wid][l16][ks * 32 + hf * 8];
      v8bf lo = *(const v8bf*)ap;
      v8bf hi = *(const v8bf*)(ap + 16);
      v16bf a;
#pragma unroll
      for (int e = 0; e < 8; ++e) { a[e] = lo[e]; a[e + 8] = hi[e]; }
#pragma unroll
      for (int n = 0; n < 4; ++n) {
        v16bf bm;
        for (int e = 0; e < 8; ++e) { bm[e] = __builtin_bit_cast(__bf16, Vs[ks * 32 + hf * 8 + e][n * 16 + l16]); bm[e + 8] = __builtin_bit_cast(__bf16, Vs[ks * 32 + 16 + hf * 8 + e][n * 16 + l16]); }
        o[n] = __builtin_amdgcn_wmma_f32_16x16x32_bf16(
            false, a, false, bm, (short)0, o[n], false, false);
      }
    }
  }

  u16* Og = O + ((size_t)b * SEQ + q0 + wid * 16) * EMB + h * HEAD_DIM;
#pragma unroll
  for (int n = 0; n < 4; ++n)
#pragma unroll
    for (int r = 0; r < 8; ++r) {
      float v = o[n][r] / l_i[r];
      Og[(size_t)(r + 8 * hf) * EMB + n * 16 + l16] = f2bf(v);
    }
}

extern "C" void kernel_launch(void* const* d_in, const int* in_sizes, int n_in,
                              void* d_out, int out_size, void* d_ws, size_t ws_size,
                              hipStream_t stream) {
  (void)in_sizes; (void)n_in; (void)out_size; (void)ws_size;
  const float* x   = (const float*)d_in[0];
  const float* ctx = (const float*)d_in[1];
  const float* q_w = (const float*)d_in[2];
  const float* q_b = (const float*)d_in[3];
  const float* k_w = (const float*)d_in[4];
  const float* k_b = (const float*)d_in[5];
  const float* v_w = (const float*)d_in[6];
  const float* v_b = (const float*)d_in[7];
  const float* o_w = (const float*)d_in[8];
  const float* o_b = (const float*)d_in[9];
  float* out = (float*)d_out;

  const size_t nX  = (size_t)BATCH * SEQ * EMB;   // 8388608
  const size_t nC  = (size_t)BATCH * CTX * DCTX;  // 2359296
  const size_t nQW = (size_t)EMB * EMB;           // 1048576
  const size_t nKW = (size_t)DCTX * EMB;          // 786432
  const size_t nKV = (size_t)BATCH * CTX * EMB;   // 3145728

  u16* w   = (u16*)d_ws;
  u16* xb  = w; w += nX;
  u16* cb  = w; w += nC;
  u16* qwb = w; w += nQW;
  u16* kwb = w; w += nKW;
  u16* vwb = w; w += nKW;
  u16* owb = w; w += nQW;
  u16* Qb  = w; w += nX;
  u16* Kb  = w; w += nKV;
  u16* Vb  = w; w += nKV;
  u16* Ob  = w; w += nX;

  auto cast = [&](const float* src, u16* dst, size_t n) {
    cast_bf16_kernel<<<dim3((unsigned)((n + 255) / 256)), 256, 0, stream>>>(src, dst, (int)n);
  };
  cast(x,   xb,  nX);
  cast(ctx, cb,  nC);
  cast(q_w, qwb, nQW);
  cast(k_w, kwb, nKW);
  cast(v_w, vwb, nKW);
  cast(o_w, owb, nQW);

  // Projections (bf16 out, fp32 accumulate)
  gemm_bias_kernel<true><<<dim3(EMB / 128, (BATCH * SEQ) / 256), 256, 0, stream>>>(
      xb, qwb, q_b, Qb, BATCH * SEQ, EMB, EMB);
  gemm_bias_kernel<true><<<dim3(EMB / 128, (BATCH * CTX) / 256), 256, 0, stream>>>(
      cb, kwb, k_b, Kb, BATCH * CTX, EMB, DCTX);
  gemm_bias_kernel<true><<<dim3(EMB / 128, (BATCH * CTX) / 256), 256, 0, stream>>>(
      cb, vwb, v_b, Vb, BATCH * CTX, EMB, DCTX);

  // Fused attention (flash-style)
  attn_kernel<<<dim3(SEQ / 128, N_HEAD, BATCH), 256, 0, stream>>>(Qb, Kb, Vb, Ob);

  // Output projection (fp32 out)
  gemm_bias_kernel<false><<<dim3(EMB / 128, (BATCH * SEQ) / 256), 256, 0, stream>>>(
      Ob, owb, o_b, out, BATCH * SEQ, EMB, EMB);
}
